// GraphEncoder_18622978195941
// MI455X (gfx1250) — matrix operand order repaired and checked by reading, not compiled, not run on hardware
//
#include <hip/hip_runtime.h>

typedef __attribute__((ext_vector_type(16))) _Float16 v16h;
typedef __attribute__((ext_vector_type(8)))  _Float16 v8h;
typedef __attribute__((ext_vector_type(8)))  float    v8f;

#define NEG_SLOPE 0.2f
#define EPS_BN 1e-5f

// ---------------- utility kernels ----------------

__global__ void k_zero_f32(float* __restrict__ p, long long n) {
  long long t = (long long)blockIdx.x * blockDim.x + threadIdx.x;
  long long stride = (long long)gridDim.x * blockDim.x;
  for (; t < n; t += stride) p[t] = 0.0f;
}

__global__ void k_cvt_half(const float* __restrict__ s, _Float16* __restrict__ d, long long n) {
  long long t = (long long)blockIdx.x * blockDim.x + threadIdx.x;
  long long stride = (long long)gridDim.x * blockDim.x;
  for (; t < n; t += stride) d[t] = (_Float16)s[t];
}

// Pack W [K x 128] (f32, row-major) into per-wave B fragments (f16):
// fragment index = ((ktile*8 + ctile)*32 + lane)*16 + j
// B 32x16 layout (16-bit): lane holds column n = ctile*16 + lane%16,
//   element j -> k = ktile*32 + (lane/16)*8 + (j%8) + (j/8)*16
__global__ void k_pack_w(const float* __restrict__ W, _Float16* __restrict__ out, int K) {
  int total = (K >> 5) * 8 * 32 * 16;
  for (int t = blockIdx.x * blockDim.x + threadIdx.x; t < total;
       t += gridDim.x * blockDim.x) {
    int j    = t & 15;
    int lane = (t >> 4) & 31;
    int ct   = (t >> 9) & 7;
    int kt   = t >> 12;
    int n = ct * 16 + (lane & 15);
    int k = kt * 32 + ((lane >> 4) << 3) + (j & 7) + ((j >> 3) << 4);
    out[t] = (_Float16)W[k * 128 + n];
  }
}

// ---------------- WMMA GEMM:  H[N,128] = X[N,K] (f16) @ Wpacked ----------------
// 256 threads = 8 waves; wave w owns column tile w (16 cols).
// Each wave preloads its K/32 B fragments once, then sweeps RT row tiles,
// so grid = ceil(N / (16*RT)).  K=128 -> 16 WMMAs/wave, K=64 -> 8.
template <int K>
__global__ __launch_bounds__(256) void k_gemm_wmma(
    const _Float16* __restrict__ X, const _Float16* __restrict__ Wp,
    float* __restrict__ Hout, int Nn) {
  constexpr int KT = K >> 5;   // k-tiles of 32
  constexpr int RT = 4;        // row tiles per wave
  const int lane  = threadIdx.x & 31;
  const int wv    = threadIdx.x >> 5;     // column tile 0..7
  const int rlo   = lane & 15;
  const int khalf = lane >> 4;
  const int col   = wv * 16 + rlo;

  // Preload B fragments (loop-invariant across row tiles).
  v16h bfrag[KT];
#pragma unroll
  for (int kt = 0; kt < KT; ++kt)
    bfrag[kt] = *(const v16h*)(Wp + (((size_t)kt * 8 + wv) * 32 + lane) * 16);

#pragma unroll
  for (int rt = 0; rt < RT; ++rt) {
    const int rowbase = (blockIdx.x * RT + rt) * 16;
    if (rowbase < Nn) {                       // uniform per wave -> scalar branch
      int row  = rowbase + rlo;
      int lrow = row < Nn ? row : (Nn - 1);   // clamp loads; EXEC all-ones for WMMA
      v8f acc = {};
#pragma unroll
      for (int kt = 0; kt < KT; ++kt) {
        // A fragment: 16-bit A 16x32 layout -> two contiguous 8xf16 loads
        const _Float16* ap = X + (size_t)lrow * K + kt * 32 + khalf * 8;
        v8h alo = *(const v8h*)(ap);
        v8h ahi = *(const v8h*)(ap + 16);
        v16h a;
#pragma unroll
        for (int i = 0; i < 8; ++i) { a[i] = alo[i]; a[i + 8] = ahi[i]; }
        acc = __builtin_amdgcn_wmma_f32_16x16x32_f16(false, a, false, bfrag[kt],
                                                     (short)0, acc, false, false);
      }
      // C/D layout: VGPR r, lane L -> row = r + (L/16)*8, col = L%16
      float* op = Hout + (size_t)(rowbase + khalf * 8) * 128 + col;
      if (rowbase + 16 <= Nn) {               // full tile: unguarded store clause
#pragma unroll
        for (int r = 0; r < 8; ++r) op[(size_t)r * 128] = acc[r];
      } else {                                // tail tile only
#pragma unroll
        for (int r = 0; r < 8; ++r)
          if (rowbase + khalf * 8 + r < Nn) op[(size_t)r * 128] = acc[r];
      }
    }
  }
}

// ---------------- attention coefficients ----------------
__global__ void k_attn(const float* __restrict__ Hf, const float* __restrict__ as,
                       const float* __restrict__ ad, float* __restrict__ asrc,
                       float* __restrict__ adst, int Nn) {
  int t = blockIdx.x * blockDim.x + threadIdx.x;
  if (t >= Nn * 2) return;
  int n = t >> 1, h = t & 1;
  const float* hp  = Hf + (size_t)n * 128 + h * 64;
  const float* asp = as + h * 64;
  const float* adp = ad + h * 64;
  float s = 0.f, d = 0.f;
#pragma unroll 8
  for (int c = 0; c < 64; ++c) { float v = hp[c]; s += v * asp[c]; d += v * adp[c]; }
  asrc[t] = s; adst[t] = d;
}

// order-preserving float<->uint encoding for atomic max
__device__ __forceinline__ unsigned fenc(float f) {
  int b = __float_as_int(f);
  return (unsigned)(b >= 0 ? (b ^ 0x80000000) : ~b);
}
__device__ __forceinline__ float fdec(unsigned k) {
  int b = (k & 0x80000000u) ? (int)(k ^ 0x80000000u) : ~(int)k;
  return __int_as_float(b);
}

__global__ void k_edge_max(const int* __restrict__ ei, int E, int Nn,
                           const float* __restrict__ asrc, const float* __restrict__ adst,
                           float* __restrict__ exbuf, unsigned* __restrict__ mkey) {
  int ET = E + Nn;
  int t = blockIdx.x * blockDim.x + threadIdx.x;
  if (t >= ET * 2) return;
  int e = t >> 1, h = t & 1;
  int s = (e < E) ? ei[e]     : (e - E);
  int d = (e < E) ? ei[E + e] : (e - E);
  float v = asrc[s * 2 + h] + adst[d * 2 + h];
  v = v > 0.f ? v : NEG_SLOPE * v;                    // leaky_relu
  exbuf[t] = v;
  atomicMax(mkey + d * 2 + h, fenc(v));
}

__global__ void k_edge_exp(const int* __restrict__ ei, int E, int Nn,
                           float* __restrict__ exbuf, const unsigned* __restrict__ mkey,
                           float* __restrict__ denom) {
  int ET = E + Nn;
  int t = blockIdx.x * blockDim.x + threadIdx.x;
  if (t >= ET * 2) return;
  int e = t >> 1, h = t & 1;
  int d = (e < E) ? ei[E + e] : (e - E);
  float x = __expf(exbuf[t] - fdec(mkey[d * 2 + h]));
  exbuf[t] = x;
  atomicAdd(denom + d * 2 + h, x);
}

__global__ void k_edge_scatter(const int* __restrict__ ei, int E, int Nn,
                               const float* __restrict__ exbuf, const float* __restrict__ denom,
                               const float* __restrict__ Hf, float* __restrict__ agg) {
  int ET = E + Nn;
  long long t = (long long)blockIdx.x * blockDim.x + threadIdx.x;
  if (t >= (long long)ET * 128) return;
  int e = (int)(t >> 7);
  int c = (int)(t & 127);
  int h = c >> 6;
  int s = (e < E) ? ei[e]     : (e - E);
  int d = (e < E) ? ei[E + e] : (e - E);
  float alpha = exbuf[e * 2 + h] / (denom[d * 2 + h] + 1e-16f);
  atomicAdd(agg + (size_t)d * 128 + c, alpha * Hf[(size_t)s * 128 + c]);
}

// head mean (H=2) + bias
__global__ void k_node_bias(const float* __restrict__ agg, const float* __restrict__ bias,
                            float* __restrict__ hnode, int Nn) {
  int t = blockIdx.x * blockDim.x + threadIdx.x;
  if (t >= Nn * 64) return;
  int n = t >> 6, c = t & 63;
  hnode[t] = 0.5f * (agg[(size_t)n * 128 + c] + agg[(size_t)n * 128 + 64 + c]) + bias[c];
}

// BatchNorm training stats: one block per channel
__global__ __launch_bounds__(256) void k_bn_stats(const float* __restrict__ hnode,
                                                  const float* __restrict__ g,
                                                  const float* __restrict__ be,
                                                  float* __restrict__ stats, int Nn) {
  __shared__ float ssum[256], ssq[256];
  int c = blockIdx.x;
  float s = 0.f, q = 0.f;
  for (int n = threadIdx.x; n < Nn; n += 256) {
    float v = hnode[(size_t)n * 64 + c];
    s += v; q += v * v;
  }
  ssum[threadIdx.x] = s; ssq[threadIdx.x] = q;
  __syncthreads();
  for (int o = 128; o > 0; o >>= 1) {
    if (threadIdx.x < o) {
      ssum[threadIdx.x] += ssum[threadIdx.x + o];
      ssq[threadIdx.x]  += ssq[threadIdx.x + o];
    }
    __syncthreads();
  }
  if (threadIdx.x == 0) {
    float mean = ssum[0] / Nn;
    float var  = ssq[0] / Nn - mean * mean;
    float sc = g[c] * rsqrtf(var + EPS_BN);
    stats[c]      = sc;
    stats[64 + c] = be[c] - mean * sc;
  }
}

__global__ void k_bn_apply(float* __restrict__ hnode, const float* __restrict__ stats,
                           _Float16* __restrict__ xh, int Nn) {
  int t = blockIdx.x * blockDim.x + threadIdx.x;
  if (t >= Nn * 64) return;
  int c = t & 63;
  float y = fmaxf(hnode[t] * stats[c] + stats[64 + c], 0.0f);
  hnode[t] = y;
  xh[t] = (_Float16)y;
}

__global__ void k_pool(const float* __restrict__ hnode, const int* __restrict__ batch,
                       float* __restrict__ psum, float* __restrict__ pcnt, int Nn) {
  int t = blockIdx.x * blockDim.x + threadIdx.x;
  if (t >= Nn * 64) return;
  int n = t >> 6, c = t & 63;
  int b = batch[n];
  atomicAdd(psum + (size_t)b * 64 + c, hnode[t]);
  if (c == 0) atomicAdd(pcnt + b, 1.0f);
}

__global__ void k_fc(const float* __restrict__ psum, const float* __restrict__ pcnt,
                     const float* __restrict__ fw, const float* __restrict__ fb,
                     float* __restrict__ out, int G, int NH) {
  int t = blockIdx.x * blockDim.x + threadIdx.x;
  if (t >= G * NH) return;
  int g = t / NH, o = t % NH;
  float cnt = fmaxf(pcnt[g], 1.0f);
  float acc = fb[o];
#pragma unroll 8
  for (int c = 0; c < 64; ++c) acc += (psum[(size_t)g * 64 + c] / cnt) * fw[c * NH + o];
  out[t] = acc;
}

// ---------------- host launch ----------------
extern "C" void kernel_launch(void* const* d_in, const int* in_sizes, int n_in,
                              void* d_out, int out_size, void* d_ws, size_t ws_size,
                              hipStream_t stream) {
  const float* x     = (const float*)d_in[0];
  const int*   ei    = (const int*)d_in[1];
  const int*   batch = (const int*)d_in[2];
  const float* Wl[3]  = {(const float*)d_in[3],  (const float*)d_in[9],  (const float*)d_in[15]};
  const float* Asl[3] = {(const float*)d_in[4],  (const float*)d_in[10], (const float*)d_in[16]};
  const float* Adl[3] = {(const float*)d_in[5],  (const float*)d_in[11], (const float*)d_in[17]};
  const float* Bl[3]  = {(const float*)d_in[6],  (const float*)d_in[12], (const float*)d_in[18]};
  const float* Gl[3]  = {(const float*)d_in[7],  (const float*)d_in[13], (const float*)d_in[19]};
  const float* Bel[3] = {(const float*)d_in[8],  (const float*)d_in[14], (const float*)d_in[20]};
  const float* fc_w = (const float*)d_in[21];
  const float* fc_b = (const float*)d_in[22];

  const int Nn = in_sizes[0] / 128;
  const int E  = in_sizes[1] / 2;
  const int NH = in_sizes[22];
  const int G  = out_size / NH;
  const int ET = E + Nn;

  // carve workspace (256-B aligned)
  char* w = (char*)d_ws;
  auto take = [&](size_t bytes) -> void* {
    void* r = (void*)w;
    w += (bytes + 255) & ~(size_t)255;
    return r;
  };
  float*     hfeat = (float*)take((size_t)Nn * 128 * 4);
  float*     agg   = (float*)take((size_t)Nn * 128 * 4);
  _Float16*  xh    = (_Float16*)take((size_t)Nn * 128 * 2);
  float*     asrc  = (float*)take((size_t)Nn * 2 * 4);
  float*     adst  = (float*)take((size_t)Nn * 2 * 4);
  unsigned*  mkey  = (unsigned*)take((size_t)Nn * 2 * 4);
  float*     denom = (float*)take((size_t)Nn * 2 * 4);
  float*     exbuf = (float*)take((size_t)ET * 2 * 4);
  float*     hnode = (float*)take((size_t)Nn * 64 * 4);
  float*     stats = (float*)take(128 * 4);
  float*     psum  = (float*)take((size_t)G * 64 * 4);
  float*     pcnt  = (float*)take((size_t)G * 4);
  _Float16*  wp0   = (_Float16*)take(16384 * 2);
  _Float16*  wp1   = (_Float16*)take(8192 * 2);
  _Float16*  wp2   = (_Float16*)take(8192 * 2);
  _Float16*  wps[3] = {wp0, wp1, wp2};
  (void)ws_size; (void)n_in;

  // pack weights + convert input features to f16
  k_pack_w<<<64, 256, 0, stream>>>(Wl[0], wp0, 128);
  k_pack_w<<<32, 256, 0, stream>>>(Wl[1], wp1, 64);
  k_pack_w<<<32, 256, 0, stream>>>(Wl[2], wp2, 64);
  k_cvt_half<<<1024, 256, 0, stream>>>(x, xh, (long long)Nn * 128);

  const int gemmBlocks = (Nn + 63) / 64;   // 4 row tiles of 16 per block
  for (int l = 0; l < 3; ++l) {
    if (l == 0)
      k_gemm_wmma<128><<<gemmBlocks, 256, 0, stream>>>(xh, wps[l], hfeat, Nn);
    else
      k_gemm_wmma<64><<<gemmBlocks, 256, 0, stream>>>(xh, wps[l], hfeat, Nn);

    k_attn<<<(Nn * 2 + 255) / 256, 256, 0, stream>>>(hfeat, Asl[l], Adl[l], asrc, adst, Nn);

    k_zero_f32<<<2048, 256, 0, stream>>>(agg, (long long)Nn * 128);
    k_zero_f32<<<256, 256, 0, stream>>>((float*)mkey, (long long)Nn * 2);
    k_zero_f32<<<256, 256, 0, stream>>>(denom, (long long)Nn * 2);

    int egrid = (ET * 2 + 255) / 256;
    k_edge_max<<<egrid, 256, 0, stream>>>(ei, E, Nn, asrc, adst, exbuf, mkey);
    k_edge_exp<<<egrid, 256, 0, stream>>>(ei, E, Nn, exbuf, mkey, denom);
    long long sthreads = (long long)ET * 128;
    k_edge_scatter<<<(unsigned)((sthreads + 255) / 256), 256, 0, stream>>>(
        ei, E, Nn, exbuf, denom, hfeat, agg);

    k_node_bias<<<(Nn * 64 + 255) / 256, 256, 0, stream>>>(agg, Bl[l], hnode, Nn);
    k_bn_stats<<<64, 256, 0, stream>>>(hnode, Gl[l], Bel[l], stats, Nn);
    k_bn_apply<<<(Nn * 64 + 255) / 256, 256, 0, stream>>>(hnode, stats, xh, Nn);
  }

  k_zero_f32<<<64, 256, 0, stream>>>(psum, (long long)G * 64);
  k_zero_f32<<<4, 256, 0, stream>>>(pcnt, (long long)G);
  k_pool<<<(Nn * 64 + 255) / 256, 256, 0, stream>>>(hnode, batch, psum, pcnt, Nn);
  k_fc<<<(G * NH + 255) / 256, 256, 0, stream>>>(psum, pcnt, fc_w, fc_b, (float*)d_out, G, NH);
}
